// HermitianGauntConvolution_74723841016351
// MI455X (gfx1250) — compile-verified, matrix operand order checked by reading, not run on hardware
//
#include <hip/hip_runtime.h>
#include <math.h>

typedef __attribute__((ext_vector_type(16))) _Float16 v16h;
typedef __attribute__((ext_vector_type(8)))  float    v8f;

#define MULC 16
#define DIMC 9
#define GRIDC 81
#define NFRAG 18   // 4 (W1) + 8 (W2) + 6 (W3) B-fragments

// ---------------------------------------------------------------------------
// Setup kernel 1: fixed linear maps Gx[9][81], Gf[9][81], P[81][9]
// reproducing:  einsum(c,U) -> pad(2) -> ifftshift -> [:, :5] -> irfft2 (Gx,Gf)
// and:          rfft2 -> einsum with Vout -> real                       (P)
// ---------------------------------------------------------------------------
__global__ void k_basis(const float* __restrict__ Uxr, const float* __restrict__ Uxi,
                        const float* __restrict__ Ufr, const float* __restrict__ Ufi,
                        const float* __restrict__ Vr,  const float* __restrict__ Vi,
                        float* __restrict__ Gx, float* __restrict__ Gf,
                        float* __restrict__ P) {
  int g = threadIdx.x;
  if (g >= GRIDC) return;
  int i = g / 9, j = g % 9;
  const float W9 = 6.28318530717958647692f / 9.0f;
  for (int d = 0; d < 9; ++d) {
    float ax = 0.f, af = 0.f;
    // 5x5 block entry (u5,v5) lands (after pad+ifftshift) at row r=(u5+7)%9,
    // col cc=(v5+7)%9; [:, :5] keeps cc<5 -> only v5=2,3,4 (cc=0,1,2) survive.
    for (int u5 = 0; u5 < 5; ++u5) {
      int r = (u5 + 7) % 9;
      for (int v5 = 2; v5 < 5; ++v5) {
        int   cc = v5 - 2;
        float w  = (cc == 0) ? (1.0f / 81.0f) : (2.0f / 81.0f);
        float th = W9 * (float)(r * i + cc * j);
        float c  = cosf(th), s = sinf(th);
        int idx = (d * 5 + u5) * 5 + v5;
        ax += w * (Uxr[idx] * c - Uxi[idx] * s);
        af += w * (Ufr[idx] * c - Ufi[idx] * s);
      }
    }
    Gx[d * GRIDC + g] = ax;
    Gf[d * GRIDC + g] = af;
    float p = 0.f;
    for (int u = 0; u < 9; ++u)
      for (int v = 0; v < 5; ++v) {
        float ph = W9 * (float)(u * i + v * j);
        int idx = (u * 5 + v) * 9 + d;
        p += Vr[idx] * cosf(ph) + Vi[idx] * sinf(ph);
      }
    P[g * 9 + d] = p;
  }
}

// ---------------------------------------------------------------------------
// Setup kernel 2: B[d][a][b] = sum_g Gx[a,g]*Gf[b,g]*P[g,d]   (729 entries)
// ---------------------------------------------------------------------------
__global__ void k_bten(const float* __restrict__ Gx, const float* __restrict__ Gf,
                       const float* __restrict__ P, float* __restrict__ B) {
  int t = blockIdx.x * blockDim.x + threadIdx.x;
  if (t >= 729) return;
  int d = t % 9, b = (t / 9) % 9, a = t / 81;
  float s = 0.f;
  for (int g = 0; g < GRIDC; ++g)
    s += Gx[a * GRIDC + g] * Gf[b * GRIDC + g] * P[g * 9 + d];
  B[d * 81 + a * 9 + b] = s;
}

// ---------------------------------------------------------------------------
// Setup kernel 3: pre-scale + convert W1/W2/W3 to f16, stored in B-FRAGMENT
// order: Wf[frag][lane][16 halves], so each lane reads 32 contiguous bytes.
// frag 0..3   = W1 tile t (K=0..31, rows>=8 zero), scale 1/sqrt(8)
// frag 4..11  = W2 (t,kc): 4 + t*2 + kc, scale 1/8
// frag 12..17 = W3 (t,kc): 12 + t*2 + kc, scale 1/8
// Fragment element map (mirrors A layout): N = n0+(lane&15),
// K = k0 + ((lane&16)?8:0) + (e<8 ? e : e+8)
// ---------------------------------------------------------------------------
__global__ void k_wfrag(const float* __restrict__ W1, const float* __restrict__ W2,
                        const float* __restrict__ W3, _Float16* __restrict__ Wf) {
  int n = blockIdx.x * blockDim.x + threadIdx.x;
  if (n >= NFRAG * 512) return;
  int f = n >> 9, r = n & 511, lane = r >> 4, e = r & 15;
  int Nl = lane & 15;
  int Kl = ((lane & 16) ? 8 : 0) + (e < 8 ? e : e + 8);
  float v;
  if (f < 4) {
    int N = f * 16 + Nl;
    v = (Kl < 8) ? W1[Kl * 64 + N] * 0.35355339059327373f : 0.f;
  } else if (f < 12) {
    int q = f - 4, t = q >> 1, kc = q & 1;
    v = W2[(kc * 32 + Kl) * 64 + (t * 16 + Nl)] * 0.125f;
  } else {
    int q = f - 12, t = q >> 1, kc = q & 1;
    v = W3[(kc * 32 + Kl) * 48 + (t * 16 + Nl)] * 0.125f;
  }
  Wf[n] = (_Float16)v;
}

// ---------------------------------------------------------------------------
// Main kernel helpers
// ---------------------------------------------------------------------------
__device__ __forceinline__ v8f zero8() {
  v8f z;
#pragma unroll
  for (int r = 0; r < 8; ++r) z[r] = 0.f;
  return z;
}

__device__ __forceinline__ v8f wmma_f16(v16h a, v16h b, v8f c) {
  return __builtin_amdgcn_wmma_f32_16x16x32_f16(false, a, false, b, (short)0, c,
                                                false, false);
}

// ShiftedSoftPlus via raw HW base-2 transcendentals (no OCML special-case
// fixups; log argument is provably in (1,2]):
//   ssp(x) = max(x,0) + ln2*log2(1 + 2^(-|x|*log2e)) - ln2
__device__ __forceinline__ float sspf(float v) {
  const float LOG2E = 1.4426950408889634f;
  const float LN2   = 0.69314718055994531f;
  float t = __builtin_amdgcn_exp2f(fabsf(v) * -LOG2E);
  float l = __builtin_amdgcn_logf(1.0f + t);
  return fmaf(l, LN2, fmaxf(v, 0.f) - LN2);
}

// Staging store: place activation h[M][K] (K=0..63) into A-fragment layout.
// A-frag: element e of lane L holds M=L&15, K = kc*32 + ((L&16)?8:0) + (e<8?e:e+8)
__device__ __forceinline__ void stage_store(_Float16* sAh, int wave, int M,
                                            int K, _Float16 val) {
  int kc = K >> 5, kr = K & 31;
  int lanehi = (kr >> 3) & 1;
  int e = (kr & 7) + ((kr >= 16) ? 8 : 0);
  sAh[(((wave * 2 + kc) * 32) + lanehi * 16 + M) * 16 + e] = val;
}

// ---------------------------------------------------------------------------
// Main kernel: 8 waves / block, one 16-edge tile per wave.
// Phase 1: 3-layer MLP via v_wmma_f32_16x16x32_f16 (18 WMMAs / tile)
// Phase 2: bilinear Gaunt contraction + a_w factor + atomic scatter
// ---------------------------------------------------------------------------
__global__ void __launch_bounds__(256) k_main(
    const float* __restrict__ x, const float* __restrict__ fsph,
    const float* __restrict__ win, const int* __restrict__ eidx,
    const float* __restrict__ aw, const float* __restrict__ den,
    const float* __restrict__ Bt, const _Float16* __restrict__ Wf,
    float* __restrict__ out, int E) {
  __shared__ v16h sFrag[NFRAG * 32];           // 18 KB: weight fragments
  __shared__ v16h sA[8 * 2 * 32];              // 16 KB: per-wave A-frag staging
  __shared__ __align__(16) float sW[8][16 * 48];  // 24 KB: per-wave w output

  int tid = threadIdx.x;
  {
    const v16h* Wv = (const v16h*)Wf;
    for (int k = tid; k < NFRAG * 32; k += 256) sFrag[k] = Wv[k];
  }
  __syncthreads();

  int wave = tid >> 5, lane = tid & 31;
  int base = (blockIdx.x * 8 + wave) * 16;
  bool hi = (lane & 16) != 0;
  int  lm = lane & 15;
  bool tile_active = (base < E);

  _Float16* sAh = (_Float16*)sA;
  float*    wst = sW[wave];

  // ---- Layer 1: A = weight_in[16 edges x 8] (K padded to 32) ----
  v16h a1;
#pragma unroll
  for (int e = 0; e < 16; ++e) a1[e] = (_Float16)0.f;
  if (!hi && tile_active) {
    int ee = base + lm; if (ee >= E) ee = E - 1;
    const float4* wv = (const float4*)(win + (size_t)ee * 8);
    float4 w0 = wv[0], w1 = wv[1];
    a1[0] = (_Float16)w0.x; a1[1] = (_Float16)w0.y;
    a1[2] = (_Float16)w0.z; a1[3] = (_Float16)w0.w;
    a1[4] = (_Float16)w1.x; a1[5] = (_Float16)w1.y;
    a1[6] = (_Float16)w1.z; a1[7] = (_Float16)w1.w;
  }
#pragma unroll
  for (int t = 0; t < 4; ++t) {
    v8f c = wmma_f16(a1, sFrag[t * 32 + lane], zero8());
#pragma unroll
    for (int r = 0; r < 8; ++r)
      stage_store(sAh, wave, r + (hi ? 8 : 0), t * 16 + lm, (_Float16)sspf(c[r]));
  }

  // ---- Layer 2: h1[16x64] @ W2[64x64] ----
  v16h a2k0 = sA[(wave * 2 + 0) * 32 + lane];
  v16h a2k1 = sA[(wave * 2 + 1) * 32 + lane];
#pragma unroll
  for (int t = 0; t < 4; ++t) {
    v8f c = wmma_f16(a2k0, sFrag[(4 + t * 2 + 0) * 32 + lane], zero8());
    c     = wmma_f16(a2k1, sFrag[(4 + t * 2 + 1) * 32 + lane], c);
#pragma unroll
    for (int r = 0; r < 8; ++r)
      stage_store(sAh, wave, r + (hi ? 8 : 0), t * 16 + lm, (_Float16)sspf(c[r]));
  }

  // ---- Layer 3: h2[16x64] @ W3[64x48] -> w[16 x 48] f32 ----
  v16h a3k0 = sA[(wave * 2 + 0) * 32 + lane];
  v16h a3k1 = sA[(wave * 2 + 1) * 32 + lane];
#pragma unroll
  for (int t = 0; t < 3; ++t) {
    v8f c = wmma_f16(a3k0, sFrag[(12 + t * 2 + 0) * 32 + lane], zero8());
    c     = wmma_f16(a3k1, sFrag[(12 + t * 2 + 1) * 32 + lane], c);
#pragma unroll
    for (int r = 0; r < 8; ++r)
      wst[(r + (hi ? 8 : 0)) * 48 + t * 16 + lm] = c[r];
  }

  // ---- Phase 2: bilinear contraction + scatter (2 lanes per edge) ----
  int  myE     = lane >> 1;
  int  halfSel = lane & 1;
  int  eg      = base + myE;
  bool active  = (eg < E);
  int  ec      = active ? eg : 0;

  int dst = eidx[ec];
  int src = eidx[E + ec];

  float f[9];
#pragma unroll
  for (int b = 0; b < 9; ++b) f[b] = fsph[(size_t)ec * 9 + b];

  float xr[72];
  {
    const float4* xv = (const float4*)(x + ((size_t)src * MULC + halfSel * 8) * DIMC);
#pragma unroll
    for (int q = 0; q < 18; ++q) {
      float4 t4 = xv[q];
      xr[q * 4 + 0] = t4.x; xr[q * 4 + 1] = t4.y;
      xr[q * 4 + 2] = t4.z; xr[q * 4 + 3] = t4.w;
    }
  }

  float wme[24];
  {
    const float4* wv = (const float4*)(wst + myE * 48 + halfSel * 24);
#pragma unroll
    for (int q = 0; q < 6; ++q) {
      float4 t4 = wv[q];
      wme[q * 4 + 0] = t4.x; wme[q * 4 + 1] = t4.y;
      wme[q * 4 + 2] = t4.z; wme[q * 4 + 3] = t4.w;
    }
  }

  float invden = 1.0f / den[0];

  for (int d = 0; d < 9; ++d) {
    const float* Bd = Bt + d * 81;  // uniform -> scalar loads
    float Cd[9];
#pragma unroll
    for (int a = 0; a < 9; ++a) {
      float s = 0.f;
#pragma unroll
      for (int b = 0; b < 9; ++b) s = fmaf(Bd[a * 9 + b], f[b], s);
      Cd[a] = s;
    }
    float a0 = aw[d], a1w = aw[9 + d], a2w = aw[18 + d];
#pragma unroll
    for (int mq = 0; mq < 8; ++mq) {
      float acc = 0.f;
#pragma unroll
      for (int a = 0; a < 9; ++a) acc = fmaf(xr[mq * 9 + a], Cd[a], acc);
      float fac = wme[mq * 3] * a0 + wme[mq * 3 + 1] * a1w + wme[mq * 3 + 2] * a2w;
      float val = acc * fac * invden;
      if (active) {
        int m = halfSel * 8 + mq;
        atomicAdd(out + ((size_t)dst * MULC + m) * DIMC + d, val);
      }
    }
  }
}

// ---------------------------------------------------------------------------
extern "C" void kernel_launch(void* const* d_in, const int* in_sizes, int n_in,
                              void* d_out, int out_size, void* d_ws, size_t ws_size,
                              hipStream_t stream) {
  const float* x    = (const float*)d_in[0];
  const float* fsph = (const float*)d_in[1];
  const float* win  = (const float*)d_in[2];
  const int*   eidx = (const int*)d_in[3];
  const float* Uxr  = (const float*)d_in[4];
  const float* Uxi  = (const float*)d_in[5];
  const float* Ufr  = (const float*)d_in[6];
  const float* Ufi  = (const float*)d_in[7];
  const float* Vr   = (const float*)d_in[8];
  const float* Vi   = (const float*)d_in[9];
  const float* W1   = (const float*)d_in[10];
  const float* W2   = (const float*)d_in[11];
  const float* W3   = (const float*)d_in[12];
  const float* aw   = (const float*)d_in[13];
  const float* den  = (const float*)d_in[14];

  int E = in_sizes[1] / DIMC;            // filter_sph is [E, 9]

  // workspace layout (Wf first: keeps 32B alignment for v16h loads)
  char* wsp = (char*)d_ws;
  _Float16* Wfrag = (_Float16*)wsp;   wsp += NFRAG * 512 * sizeof(_Float16);
  float* Gx = (float*)wsp;            wsp += GRIDC * 9 * sizeof(float);
  float* Gf = (float*)wsp;            wsp += GRIDC * 9 * sizeof(float);
  float* P  = (float*)wsp;            wsp += GRIDC * 9 * sizeof(float);
  float* B  = (float*)wsp;            wsp += 729 * sizeof(float);

  k_basis<<<1, 128, 0, stream>>>(Uxr, Uxi, Ufr, Ufi, Vr, Vi, Gx, Gf, P);
  k_bten<<<3, 256, 0, stream>>>(Gx, Gf, P, B);
  k_wfrag<<<(NFRAG * 512 + 255) / 256, 256, 0, stream>>>(W1, W2, W3, Wfrag);

  hipMemsetAsync(d_out, 0, (size_t)out_size * sizeof(float), stream);

  int tiles  = (E + 15) / 16;
  int blocks = (tiles + 7) / 8;
  k_main<<<blocks, 256, 0, stream>>>(x, fsph, win, eidx, aw, den, B, Wfrag,
                                     (float*)d_out, E);
}
